// GAT_3032246911641
// MI455X (gfx1250) — matrix operand order repaired and checked by reading, not compiled, not run on hardware
//
#include <hip/hip_runtime.h>
#include <cstdint>

#define NN 100000
#define NE 1000000

typedef __attribute__((ext_vector_type(16))) __bf16 v16bf;
typedef __attribute__((ext_vector_type(8)))  __bf16 v8bf;
typedef __attribute__((ext_vector_type(8)))  float  v8f;

// ---------- atomic helpers (lower to global_atomic_add_f32 / max_i32 / min_u32) ----------
__device__ __forceinline__ void atomAddF(float* p, float v) {
    __hip_atomic_fetch_add(p, v, __ATOMIC_RELAXED, __HIP_MEMORY_SCOPE_AGENT);
}
__device__ __forceinline__ void atomMaxF(float* p, float v) {
    // monotonic int-bits trick: works for mixed signs given -inf init
    if (__float_as_int(v) >= 0) atomicMax((int*)p, __float_as_int(v));
    else                        atomicMin((unsigned int*)p, __float_as_uint(v));
}

// ---------- fill ----------
__global__ void k_fill(float* __restrict__ p, float v, size_t n) {
    size_t i = (size_t)blockIdx.x * blockDim.x + threadIdx.x;
    size_t stride = (size_t)gridDim.x * blockDim.x;
    for (; i < n; i += stride) p[i] = v;
}

// ---------- W[k,c] (row-major fp32) -> Wt[c,k] (bf16, col-major for B-operand) ----------
__global__ void k_convW(const float* __restrict__ W, __bf16* __restrict__ Wt, int K, int OUT) {
    int i = blockIdx.x * blockDim.x + threadIdx.x;
    if (i >= K * OUT) return;
    int k = i / OUT, c = i % OUT;
    Wt[(size_t)c * K + k] = (__bf16)W[i];
}

// ---------- C[N,OUT] = A[N,128](fp32->bf16 via LDS) @ B(from Wt[OUT,128] bf16), WMMA f32 accum ----
// block = 32 * (OUT/16) threads; one wave per 16-wide column tile; grid.x = N/16.
// A tile (16x128) staged once per block in LDS as bf16 with padded stride (bank rotation).
#define GK 128
#define LDA 136   // 128 + 8 bf16 pad: 272B row stride -> 4-bank rotation per row
__global__ void k_gemm(const float* __restrict__ A, const __bf16* __restrict__ Bt,
                       float* __restrict__ C, int OUT) {
    __shared__ __bf16 As[16 * LDA];

    // ---- cooperative A-tile load: 16 rows x 128 cols, 8 elems per iteration ----
    for (int i = threadIdx.x; i < 16 * (GK / 8); i += blockDim.x) {
        int r = i >> 4;              // row 0..15
        int c = (i & 15) * 8;        // col 0..120
        const float* g = A + (size_t)(blockIdx.x * 16 + r) * GK + c;
        float4 f0 = *(const float4*)(g);
        float4 f1 = *(const float4*)(g + 4);
        v8bf t;
        t[0]=(__bf16)f0.x; t[1]=(__bf16)f0.y; t[2]=(__bf16)f0.z; t[3]=(__bf16)f0.w;
        t[4]=(__bf16)f1.x; t[5]=(__bf16)f1.y; t[6]=(__bf16)f1.z; t[7]=(__bf16)f1.w;
        *(v8bf*)(As + r * LDA + c) = t;
    }
    __syncthreads();

    const int lane = threadIdx.x & 31;
    const int wave = threadIdx.x >> 5;
    const int l    = lane & 15;
    const int half = lane >> 4;            // lane group 0/1
    const int colB = wave * 16;

    const __bf16* arow = As + l * LDA;
    const __bf16* brow = Bt + (size_t)(colB + l) * GK;

    v8f acc = {};
    #pragma unroll
    for (int kk = 0; kk < GK; kk += 32) {
        // A 16x32 bf16 layout: lanes 0-15 row M=l hold K = kk+{0..7,16..23}; lanes 16-31 hold {8..15,24..31}
        v8bf a0 = *(const v8bf*)(arow + kk + half * 8);
        v8bf a1 = *(const v8bf*)(arow + kk + 16 + half * 8);
        v16bf a;
        #pragma unroll
        for (int i = 0; i < 8; ++i) { a[i] = a0[i]; a[8 + i] = a1[i]; }

        // B 32x16 bf16: lanes 0-15 col N=l hold K = kk+{0..7,16..23}; lanes 16-31 hold {8..15,24..31}
        v8bf b0 = *(const v8bf*)(brow + kk + half * 8);
        v8bf b1 = *(const v8bf*)(brow + kk + 16 + half * 8);
        v16bf b;
        #pragma unroll
        for (int i = 0; i < 8; ++i) { b[i] = b0[i]; b[8 + i] = b1[i]; }

        acc = __builtin_amdgcn_wmma_f32_16x16x32_bf16(
            false, a, false, b, (short)0, acc, false, false);
    }
    // C/D layout: lanes 0-15: VGPR r -> M=r, N=lane; lanes 16-31: M=8+r, N=lane-16
    #pragma unroll
    for (int r = 0; r < 8; ++r) {
        C[(size_t)(blockIdx.x * 16 + half * 8 + r) * OUT + colB + l] = acc[r];
    }
}

// ---------- per (node,head): el/er dots + init m=-inf, s=0 ----------
__global__ void k_scores(const float* __restrict__ feat, const float* __restrict__ al,
                         const float* __restrict__ ar, float* __restrict__ el,
                         float* __restrict__ er, float* __restrict__ m,
                         float* __restrict__ s, int NH, int H, int D) {
    int tid = blockIdx.x * blockDim.x + threadIdx.x;
    if (tid >= NH) return;
    int h = tid % H;
    const float* f = feat + (size_t)(tid / H) * (H * D) + h * D;
    float a = 0.f, b = 0.f;
    for (int d = 0; d < D; ++d) { float x = f[d]; a += x * al[h * D + d]; b += x * ar[h * D + d]; }
    el[tid] = a; er[tid] = b; m[tid] = -3.0e38f; s[tid] = 0.f;
}

// ---------- edge pass 1: e = lrelu(el[src]+er[dst]); stash e; atomic max into m[dst] ----------
__global__ void k_edge_max(const int* __restrict__ src, const int* __restrict__ dst,
                           const float* __restrict__ el, const float* __restrict__ er,
                           float* __restrict__ ebuf, float* __restrict__ m, int EH, int H) {
    int tid = blockIdx.x * blockDim.x + threadIdx.x;
    if (tid >= EH) return;
    int h = tid % H, e = tid / H;
    float v = el[src[e] * H + h] + er[dst[e] * H + h];
    v = v > 0.f ? v : 0.2f * v;
    ebuf[tid] = v;
    atomMaxF(&m[dst[e] * H + h], v);
}

// ---------- edge pass 2: ex = exp(e - m[dst]); atomic sum into s[dst] ----------
__global__ void k_edge_exp(const int* __restrict__ dst, float* __restrict__ ebuf,
                           const float* __restrict__ m, float* __restrict__ s, int EH, int H) {
    int tid = blockIdx.x * blockDim.x + threadIdx.x;
    if (tid >= EH) return;
    int h = tid % H, e = tid / H;
    int di = dst[e] * H + h;
    float v = __expf(ebuf[tid] - m[di]);
    ebuf[tid] = v;
    atomAddF(&s[di], v);
}

// ---------- edge pass 3: out[dst] += feat[src] * alpha ; OUTF/4 lanes per edge, float4 gather ----------
__global__ void k_aggregate(const int* __restrict__ src, const int* __restrict__ dst,
                            const float* __restrict__ feat, const float* __restrict__ ebuf,
                            const float* __restrict__ s, float* __restrict__ out,
                            int E, int OUTF, int H) {
    int tpe = OUTF >> 2;
    int gid = blockIdx.x * blockDim.x + threadIdx.x;
    int e = gid / tpe;
    if (e >= E) return;
    int c = (gid % tpe) * 4;
    int h = c >> 5;                         // D = 32
    int sn = src[e], dn = dst[e];
    float alpha = ebuf[(size_t)e * H + h] / s[dn * H + h];
    float4 f = *(const float4*)(feat + (size_t)sn * OUTF + c);
    float* o = out + (size_t)dn * OUTF + c;
    atomAddF(o + 0, f.x * alpha);
    atomAddF(o + 1, f.y * alpha);
    atomAddF(o + 2, f.z * alpha);
    atomAddF(o + 3, f.w * alpha);
}

// ---------- epilogue: out = act(agg + bias [+ resid]) ----------
__global__ void k_epilogue(const float* __restrict__ agg, const float* __restrict__ bias,
                           const float* __restrict__ resid, float* __restrict__ out,
                           int NF, int F, int doElu) {
    int tid = blockIdx.x * blockDim.x + threadIdx.x;
    if (tid >= NF) return;
    int c = tid % F;
    float v = agg[tid] + bias[c];
    if (resid) v += resid[tid];
    if (doElu) v = v > 0.f ? v : (__expf(v) - 1.f);
    out[tid] = v;
}

// ---------- classifier: out[n,c] = h3[n,:] . Wfc[:,c] + bfc[c] ; one block per node ----------
__global__ void k_fc(const float* __restrict__ h, const float* __restrict__ Wfc,
                     const float* __restrict__ bfc, float* __restrict__ out, int N) {
    int n = blockIdx.x, c = threadIdx.x;
    if (c >= 40) return;
    const float* hr = h + (size_t)n * 32;
    float acc = bfc[c];
    #pragma unroll
    for (int k = 0; k < 32; ++k) acc += hr[k] * Wfc[k * 40 + c];
    out[(size_t)n * 40 + c] = acc;
}

extern "C" void kernel_launch(void* const* d_in, const int* in_sizes, int n_in,
                              void* d_out, int out_size, void* d_ws, size_t ws_size,
                              hipStream_t stream) {
    const float* x    = (const float*)d_in[0];
    const int*   src  = (const int*)d_in[1];
    const int*   dst  = (const int*)d_in[2];
    const float* W1   = (const float*)d_in[3];
    const float* al1  = (const float*)d_in[4];
    const float* ar1  = (const float*)d_in[5];
    const float* b1   = (const float*)d_in[6];
    const float* W2   = (const float*)d_in[7];
    const float* al2  = (const float*)d_in[8];
    const float* ar2  = (const float*)d_in[9];
    const float* b2   = (const float*)d_in[10];
    const float* W3   = (const float*)d_in[11];
    const float* al3  = (const float*)d_in[12];
    const float* ar3  = (const float*)d_in[13];
    const float* b3   = (const float*)d_in[14];
    const float* res3 = (const float*)d_in[15];
    const float* Wfc  = (const float*)d_in[16];
    const float* bfc  = (const float*)d_in[17];

    const size_t N = NN, E = NE;
    float*  feat = (float*)d_ws;                       // N*128
    float*  outb = feat + N * 128;                     // N*128 (accumulator)
    float*  hbuf = outb + N * 128;                     // N*128 (h1 / layer-3 scratch)
    float*  el   = hbuf + N * 128;                     // N*4
    float*  er   = el + N * 4;
    float*  mb   = er + N * 4;
    float*  sb   = mb + N * 4;
    float*  ebuf = sb + N * 4;                         // E*4
    __bf16* wtb  = (__bf16*)(ebuf + E * 4);            // 128*128 bf16

    const int TB = 256;
    #define GRD(n) ((int)(((n) + TB - 1) / TB))

    // ---------------- layer 1: in = x, no residual, ELU -> hbuf ----------------
    k_convW<<<GRD(128 * 128), TB, 0, stream>>>(W1, wtb, 128, 128);
    k_gemm<<<(int)(N / 16), 256, 0, stream>>>(x, wtb, feat, 128);
    k_scores<<<GRD(N * 4), TB, 0, stream>>>(feat, al1, ar1, el, er, mb, sb, (int)(N * 4), 4, 32);
    k_edge_max<<<GRD(E * 4), TB, 0, stream>>>(src, dst, el, er, ebuf, mb, (int)(E * 4), 4);
    k_edge_exp<<<GRD(E * 4), TB, 0, stream>>>(dst, ebuf, mb, sb, (int)(E * 4), 4);
    k_fill<<<4096, TB, 0, stream>>>(outb, 0.f, N * 128);
    k_aggregate<<<GRD(E * 32), TB, 0, stream>>>(src, dst, feat, ebuf, sb, outb, (int)E, 128, 4);
    k_epilogue<<<GRD(N * 128), TB, 0, stream>>>(outb, b1, nullptr, hbuf, (int)(N * 128), 128, 1);

    // ---------------- layer 2: in = hbuf, identity residual, ELU -> feat ----------------
    k_convW<<<GRD(128 * 128), TB, 0, stream>>>(W2, wtb, 128, 128);
    k_gemm<<<(int)(N / 16), 256, 0, stream>>>(hbuf, wtb, feat, 128);
    k_scores<<<GRD(N * 4), TB, 0, stream>>>(feat, al2, ar2, el, er, mb, sb, (int)(N * 4), 4, 32);
    k_edge_max<<<GRD(E * 4), TB, 0, stream>>>(src, dst, el, er, ebuf, mb, (int)(E * 4), 4);
    k_edge_exp<<<GRD(E * 4), TB, 0, stream>>>(dst, ebuf, mb, sb, (int)(E * 4), 4);
    k_fill<<<4096, TB, 0, stream>>>(outb, 0.f, N * 128);
    k_aggregate<<<GRD(E * 32), TB, 0, stream>>>(src, dst, feat, ebuf, sb, outb, (int)E, 128, 4);
    k_epilogue<<<GRD(N * 128), TB, 0, stream>>>(outb, b2, hbuf, feat, (int)(N * 128), 128, 1);
    // h2 now lives in feat

    // ---------------- layer 3: in = feat (h2), projected residual, no activation ----------------
    const size_t N32 = N * 32;
    float* feat3 = hbuf;             // N*32
    float* agg3  = hbuf + N32;       // N*32
    float* res3v = hbuf + 2 * N32;   // N*32
    float* h3    = hbuf + 3 * N32;   // N*32

    k_convW<<<GRD(128 * 32), TB, 0, stream>>>(W3, wtb, 128, 32);
    k_gemm<<<(int)(N / 16), 64, 0, stream>>>(feat, wtb, feat3, 32);
    k_scores<<<GRD(N), TB, 0, stream>>>(feat3, al3, ar3, el, er, mb, sb, (int)N, 1, 32);
    k_edge_max<<<GRD(E), TB, 0, stream>>>(src, dst, el, er, ebuf, mb, (int)E, 1);
    k_edge_exp<<<GRD(E), TB, 0, stream>>>(dst, ebuf, mb, sb, (int)E, 1);
    k_fill<<<4096, TB, 0, stream>>>(agg3, 0.f, N32);
    k_aggregate<<<GRD(E * 8), TB, 0, stream>>>(src, dst, feat3, ebuf, sb, agg3, (int)E, 32, 1);
    k_convW<<<GRD(128 * 32), TB, 0, stream>>>(res3, wtb, 128, 32);
    k_gemm<<<(int)(N / 16), 64, 0, stream>>>(feat, wtb, res3v, 32);
    k_epilogue<<<GRD(N * 32), TB, 0, stream>>>(agg3, b3, res3v, h3, (int)(N * 32), 32, 0);

    // ---------------- classifier ----------------
    k_fc<<<(int)N, 64, 0, stream>>>(h3, Wfc, bfc, (float*)d_out, (int)N);
    #undef GRD
}
